// FullAttention_42880953483611
// MI455X (gfx1250) — matrix operand order repaired and checked by reading, not compiled, not run on hardware
//
#include <hip/hip_runtime.h>
#include <stdint.h>

// MI455X / gfx1250 flash-attention forward, wave32, bf16 WMMA, fp32 accumulate.
// Pre-pass converts K -> bf16 [N,S,H,D] and V -> bf16 transposed [N,H,D,S] in
// d_ws. The main kernel stages K/V 32-key tiles into LDS cooperatively with
// GLOBAL_LOAD_ASYNC_TO_LDS_B128 (double-buffered, ASYNCcnt + barrier handoff),
// and S^T = K x Q^T keeps softmaxed probabilities in WMMA A-layout.

typedef __attribute__((ext_vector_type(16))) __bf16       v16bf;
typedef __attribute__((ext_vector_type(16))) short        v16s;
typedef __attribute__((ext_vector_type(8)))  short        v8s;
typedef __attribute__((ext_vector_type(8)))  float        v8f;
typedef __attribute__((ext_vector_type(4)))  float        v4f;
typedef __attribute__((ext_vector_type(4)))  int          v4i;
typedef __attribute__((ext_vector_type(8)))  unsigned int v8u;
typedef __attribute__((ext_vector_type(2)))  unsigned int v2u;

constexpr int N_ = 4, L_ = 2048, S_ = 2048, H_ = 8, D_ = 64;
constexpr int ROW   = H_ * D_;            // 512 elements between consecutive l / s
constexpr int WAVES = 8;                  // waves per block
constexpr size_t KBF_BYTES = (size_t)N_ * S_ * H_ * D_ * 2;   // 8 MiB
constexpr size_t WS_NEEDED = 2 * KBF_BYTES;                   // K-bf16 + V^T-bf16

#define GLAS  __attribute__((address_space(1)))
#define LDSAS __attribute__((address_space(3)))

#if defined(__has_builtin)
#if __has_builtin(__builtin_amdgcn_global_load_async_to_lds_b128)
#define HAVE_ASYNC_LDS 1
#endif
#endif

#if defined(HAVE_ASYNC_LDS)
#if __has_builtin(__builtin_amdgcn_s_wait_asynccnt)
#define WAIT_ASYNC() __builtin_amdgcn_s_wait_asynccnt(0)
#else
#define WAIT_ASYNC() asm volatile("s_wait_asynccnt 0" ::: "memory")
#endif
#else
#define WAIT_ASYNC()
#endif

__device__ __forceinline__ uint32_t fbits(float x) {
    union { float f; uint32_t u; } v; v.f = x; return v.u;
}
__device__ __forceinline__ unsigned short bf16rne(float x) {
    uint32_t u = fbits(x);
    u += 0x7fffu + ((u >> 16) & 1u);
    return (unsigned short)(u >> 16);
}
// pack two fp32 -> one dword of two bf16 (truncate): single v_perm_b32
__device__ __forceinline__ uint32_t pk2bf(float lo, float hi) {
    return __builtin_amdgcn_perm(fbits(hi), fbits(lo), 0x07060302u);
}

// 16-byte chunk: global -> LDS (async if available)
__device__ __forceinline__ void cp16(const short* g, short* l) {
#if defined(HAVE_ASYNC_LDS)
    __builtin_amdgcn_global_load_async_to_lds_b128(
        (GLAS v4i*)(uintptr_t)g,
        (LDSAS v4i*)(unsigned int)(uintptr_t)l, 0, 0);
#else
    *(v8s*)l = *(const v8s*)g;
#endif
}

// ---------------------------------------------------------------- pre-passes
__global__ __launch_bounds__(256)
void conv_k_bf16(const float* __restrict__ k, uint32_t* __restrict__ kbf) {
    size_t i = (size_t)blockIdx.x * 256 + threadIdx.x;   // 4 floats -> 2 dwords
    v4f f = *(const v4f*)(k + i * 4);
    v2u o;
    o.x = (uint32_t)bf16rne(f[0]) | ((uint32_t)bf16rne(f[1]) << 16);
    o.y = (uint32_t)bf16rne(f[2]) | ((uint32_t)bf16rne(f[3]) << 16);
    *(v2u*)(kbf + i * 2) = o;
}

__global__ __launch_bounds__(256)
void conv_v_bf16T(const float* __restrict__ v, uint32_t* __restrict__ vt) {
    // vt[((n*H+h)*D + d)*S + s] = bf16(v[((n*S+s)*H+h)*D + d]); 2 s per thread
    size_t i  = (size_t)blockIdx.x * 256 + threadIdx.x;
    int s2  = (int)(i % (S_ / 2));
    int rem = (int)(i / (S_ / 2));
    int d   = rem % D_;
    int nh  = rem / D_;
    int n   = nh / H_, h = nh % H_;
    size_t src = ((size_t)(n * S_ + 2 * s2) * H_ + h) * D_ + d;
    uint32_t o = (uint32_t)bf16rne(v[src]) | ((uint32_t)bf16rne(v[src + ROW]) << 16);
    vt[((size_t)nh * D_ + d) * (S_ / 2) + s2] = o;
}

// ------------------------------------------------------- main kernel (bf16 ws)
// Block = 8 waves sharing one (n,h); each wave owns a 32-row Q tile. K/V tiles
// are staged once per block into double-buffered LDS via async-to-LDS.
__global__ __launch_bounds__(WAVES * 32)
void fattn_bf16(const float* __restrict__ q,
                const short* __restrict__ kbf,    // [N,S,H,D] bf16
                const short* __restrict__ vt,     // [N,H,D,S] bf16
                const unsigned char* __restrict__ kvmask,
                float* __restrict__ out)
{
    __shared__ __align__(16) short Kb[2][32][64];   // [buf][key][d]     4 KB
    __shared__ __align__(16) short Vb[2][64][32];   // [buf][d][key]     4 KB

    const int tid  = threadIdx.x;
    const int lane = tid & 31;
    const int wave = tid >> 5;
    const int tile = blockIdx.x * WAVES + wave;   // 0 .. N*H*(L/32)-1
    constexpr int QT = L_ / 32;                   // 64 (divisible by WAVES)
    const int qt = tile % QT;
    const int nh = tile / QT;                     // uniform across block
    const int h  = nh & (H_ - 1);
    const int n  = nh / H_;
    const int qbase = qt * 32;
    const int hi  = (lane >> 4) & 1;
    const int l15 = lane & 15;

    // Q^T B-fragments [subtile][chunk]; lane -> q row l15, elem e -> d=32c+8hi+e (e<8), 32c+16+8hi+(e-8) (e>=8)
    v16bf qB[2][2];
    #pragma unroll
    for (int s = 0; s < 2; ++s) {
        const float* qrow = q + ((size_t)((n * L_ + qbase + 16 * s + l15) * H_) + h) * D_;
        #pragma unroll
        for (int c = 0; c < 2; ++c) {
            const float* p = qrow + 32 * c + 8 * hi;
            v4f f0 = *(const v4f*)(p + 0);
            v4f f1 = *(const v4f*)(p + 4);
            v4f f2 = *(const v4f*)(p + 16);
            v4f f3 = *(const v4f*)(p + 20);
            v16s t;
            #pragma unroll
            for (int e = 0; e < 4; ++e) {
                t[e]      = (short)bf16rne(f0[e]);
                t[4 + e]  = (short)bf16rne(f1[e]);
                t[8 + e]  = (short)bf16rne(f2[e]);
                t[12 + e] = (short)bf16rne(f3[e]);
            }
            qB[s][c] = __builtin_bit_cast(v16bf, t);
        }
    }

    const short* kb = kbf + ((size_t)(n * S_) * H_ + h) * D_;
    const short* vb = vt + (size_t)nh * D_ * S_;
    const unsigned char* km = kvmask + (size_t)n * S_;

    // per-thread staging slots: K chunk (row=tid/8, seg=tid%8), V chunk (d=tid/4)
    const int krow_st = tid >> 3, kseg = (tid & 7) * 8;
    const int vrow_st = tid >> 2, vseg = (tid & 3) * 8;

    v8f o[2][4] = {};                       // [subtile][d-block], 32 rows x 64 d
    float m_run[2] = { -3.0e38f, -3.0e38f };
    float l_run[2] = { 0.0f, 0.0f };
    const float sc_ = 0.125f * 1.44269504088896340736f; // 1/sqrt(D) * log2(e)

    // prologue: stage tile 0 into buffer 0
    cp16(kb + (size_t)krow_st * ROW + kseg, &Kb[0][krow_st][kseg]);
    cp16(vb + (size_t)vrow_st * S_ + vseg,  &Vb[0][vrow_st][vseg]);
    WAIT_ASYNC();
    __syncthreads();

    int cur = 0;
    for (int sb = 0; sb < S_; sb += 32) {
        const int nxt = cur ^ 1;
        if (sb + 32 < S_) {   // stream next K/V tile while computing this one
            cp16(kb + (size_t)(sb + 32 + krow_st) * ROW + kseg, &Kb[nxt][krow_st][kseg]);
            cp16(vb + (size_t)vrow_st * S_ + (sb + 32) + vseg,  &Vb[nxt][vrow_st][vseg]);
        }

        unsigned int mask32 = (unsigned int)__ballot(km[sb + lane] != 0);

        // ---- S^T = K_tile x Q^T : A = K rows from LDS (shared by subtiles) --
        v8f scv[2][2];                      // [key subtile j][q subtile s]
        #pragma unroll
        for (int j = 0; j < 2; ++j) {
            const short* krow = &Kb[cur][16 * j + l15][0];
            v16bf aF[2];
            #pragma unroll
            for (int c = 0; c < 2; ++c) {
                const int d0 = 32 * c + 8 * hi;
                v8s lo = *(const v8s*)(krow + d0);
                v8s hs = *(const v8s*)(krow + d0 + 16);
                v16s t = __builtin_shufflevector(lo, hs, 0,1,2,3,4,5,6,7,
                                                         8,9,10,11,12,13,14,15);
                aF[c] = __builtin_bit_cast(v16bf, t);
            }
            #pragma unroll
            for (int s = 0; s < 2; ++s) {
                v8f acc = {};
                acc = __builtin_amdgcn_wmma_f32_16x16x32_bf16(
                          false, aF[0], false, qB[s][0], (short)0, acc, false, false);
                acc = __builtin_amdgcn_wmma_f32_16x16x32_bf16(
                          false, aF[1], false, qB[s][1], (short)0, acc, false, false);
                scv[j][s] = acc;
            }
        }

        // ---- V B-fragments from LDS: lane -> d=16nb+l15, elem e -> key ----
        v16bf vF[4];
        #pragma unroll
        for (int nb = 0; nb < 4; ++nb) {
            const short* vrow = &Vb[cur][nb * 16 + l15][8 * hi];
            v8s lo = *(const v8s*)(vrow);
            v8s hs = *(const v8s*)(vrow + 16);
            v16s t = __builtin_shufflevector(lo, hs, 0,1,2,3,4,5,6,7,
                                                     8,9,10,11,12,13,14,15);
            vF[nb] = __builtin_bit_cast(v16bf, t);
        }

        const bool allvalid = (mask32 == 0xffffffffu);  // wave-uniform

        #pragma unroll
        for (int s = 0; s < 2; ++s) {
            // online softmax over keys: elem r -> key 16j+r+8hi, lane -> q row
            float s0[8], s1[8];
            float mb = -3.0e38f;
            #pragma unroll
            for (int r = 0; r < 8; ++r) {
                float x0 = scv[0][s][r] * sc_;
                float x1 = scv[1][s][r] * sc_;
                if (!allvalid) {
                    if (!((mask32 >> (r + 8 * hi)) & 1u))      x0 = -3.0e38f;
                    if (!((mask32 >> (16 + r + 8 * hi)) & 1u)) x1 = -3.0e38f;
                }
                s0[r] = x0; s1[r] = x1;
                mb = fmaxf(mb, fmaxf(x0, x1));
            }
            mb = fmaxf(mb, __shfl_xor(mb, 16, 32));
            const float m_new = fmaxf(m_run[s], mb);

            float corr = 1.0f;
            if (__any(mb > m_run[s])) {     // max moved: rescale O (rare late)
                corr = exp2f(m_run[s] - m_new);
                const int srcl = 8 * hi;
                #pragma unroll
                for (int r = 0; r < 8; ++r) {
                    float cr = __shfl(corr, srcl + r, 32);
                    o[s][0][r] *= cr; o[s][1][r] *= cr;
                    o[s][2][r] *= cr; o[s][3][r] *= cr;
                }
            }

            float p0[8], p1[8], lsum = 0.0f;
            #pragma unroll
            for (int r = 0; r < 8; ++r) {
                p0[r] = exp2f(s0[r] - m_new);
                p1[r] = exp2f(s1[r] - m_new);
                lsum += p0[r] + p1[r];
            }
            l_run[s] = l_run[s] * corr + lsum;
            m_run[s] = m_new;

            // P in S^T C-layout == A-layout for P x V ; pack via v_perm_b32
            v8u pv;
            #pragma unroll
            for (int w = 0; w < 4; ++w) {
                pv[w]     = pk2bf(p0[2 * w], p0[2 * w + 1]);
                pv[4 + w] = pk2bf(p1[2 * w], p1[2 * w + 1]);
            }
            v16bf pF = __builtin_bit_cast(v16bf, pv);

            #pragma unroll
            for (int nb = 0; nb < 4; ++nb)
                o[s][nb] = __builtin_amdgcn_wmma_f32_16x16x32_bf16(
                               false, pF, false, vF[nb], (short)0, o[s][nb],
                               false, false);
        }

        WAIT_ASYNC();       // own async loads for buf[nxt] complete
        __syncthreads();    // whole block done loading nxt / reading cur
        cur = nxt;
    }

    // ------------------------- finalize: divide by row sum, store -----------
    #pragma unroll
    for (int s = 0; s < 2; ++s) {
        float lt = l_run[s] + __shfl_xor(l_run[s], 16, 32);
        const int srcl = 8 * hi;
        #pragma unroll
        for (int r = 0; r < 8; ++r) {
            float inv = 1.0f / __shfl(lt, srcl + r, 32);
            float* orow = out + ((size_t)((n * L_ + qbase + 16 * s + r + 8 * hi) * H_) + h) * D_ + l15;
            orow[0]  = o[s][0][r] * inv;
            orow[16] = o[s][1][r] * inv;
            orow[32] = o[s][2][r] * inv;
            orow[48] = o[s][3][r] * inv;
        }
    }
}

// ----------------------- fallback (ws too small): fused conversion, 16 q rows
__global__ __launch_bounds__(WAVES * 32)
void fattn_fused(const float* __restrict__ q, const float* __restrict__ k,
                 const float* __restrict__ v,
                 const unsigned char* __restrict__ kvmask,
                 float* __restrict__ out)
{
    const int lane = threadIdx.x & 31;
    const int wave = threadIdx.x >> 5;
    const int tile = blockIdx.x * WAVES + wave;
    constexpr int QT = L_ / 16;
    const int qt = tile % QT;
    const int nh = tile / QT;
    const int h  = nh & (H_ - 1);
    const int n  = nh / H_;
    const int qbase = qt * 16;
    const int hi  = (lane >> 4) & 1;
    const int l15 = lane & 15;

    const float* qrow = q + ((size_t)((n * L_ + qbase + l15) * H_) + h) * D_;
    v16bf qB[2];
    #pragma unroll
    for (int c = 0; c < 2; ++c) {
        const float* p = qrow + 32 * c + 8 * hi;
        v4f f0 = *(const v4f*)(p + 0);
        v4f f1 = *(const v4f*)(p + 4);
        v4f f2 = *(const v4f*)(p + 16);
        v4f f3 = *(const v4f*)(p + 20);
        v16s t;
        #pragma unroll
        for (int e = 0; e < 4; ++e) {
            t[e] = (short)bf16rne(f0[e]);  t[4 + e]  = (short)bf16rne(f1[e]);
            t[8 + e] = (short)bf16rne(f2[e]); t[12 + e] = (short)bf16rne(f3[e]);
        }
        qB[c] = __builtin_bit_cast(v16bf, t);
    }

    const float* kb = k + ((size_t)(n * S_) * H_ + h) * D_;
    const float* vb = v + ((size_t)(n * S_) * H_ + h) * D_;
    const unsigned char* km = kvmask + (size_t)n * S_;

    v8f o0 = {}, o1 = {}, o2 = {}, o3 = {};
    float m_run = -3.0e38f, l_run = 0.0f;
    const float sc_ = 0.125f * 1.44269504088896340736f;

    for (int sb = 0; sb < S_; sb += 32) {
        unsigned int mask32 = (unsigned int)__ballot(km[sb + lane] != 0);
        v8f c0 = {}, c1 = {};
        #pragma unroll
        for (int j = 0; j < 2; ++j) {
            const float* krow = kb + (size_t)(sb + 16 * j + l15) * ROW;
            v8f acc = {};
            #pragma unroll
            for (int c = 0; c < 2; ++c) {
                const int d0 = 32 * c + 8 * hi;
                v4f a0 = *(const v4f*)(krow + d0 + 0);
                v4f a1 = *(const v4f*)(krow + d0 + 4);
                v4f a2 = *(const v4f*)(krow + d0 + 16);
                v4f a3 = *(const v4f*)(krow + d0 + 20);
                v8u tw;
                #pragma unroll
                for (int e = 0; e < 2; ++e) {
                    tw[e]     = pk2bf(a0[2 * e], a0[2 * e + 1]);
                    tw[2 + e] = pk2bf(a1[2 * e], a1[2 * e + 1]);
                    tw[4 + e] = pk2bf(a2[2 * e], a2[2 * e + 1]);
                    tw[6 + e] = pk2bf(a3[2 * e], a3[2 * e + 1]);
                }
                v16bf aF = __builtin_bit_cast(v16bf, tw);
                acc = __builtin_amdgcn_wmma_f32_16x16x32_bf16(
                          false, aF, false, qB[c], (short)0, acc, false, false);
            }
            if (j == 0) c0 = acc; else c1 = acc;
        }

        const bool allvalid = (mask32 == 0xffffffffu);
        float s0[8], s1[8], mb = -3.0e38f;
        #pragma unroll
        for (int r = 0; r < 8; ++r) {
            float x0 = c0[r] * sc_;
            float x1 = c1[r] * sc_;
            if (!allvalid) {
                if (!((mask32 >> (r + 8 * hi)) & 1u))      x0 = -3.0e38f;
                if (!((mask32 >> (16 + r + 8 * hi)) & 1u)) x1 = -3.0e38f;
            }
            s0[r] = x0; s1[r] = x1;
            mb = fmaxf(mb, fmaxf(x0, x1));
        }
        mb = fmaxf(mb, __shfl_xor(mb, 16, 32));
        const float m_new = fmaxf(m_run, mb);
        const float corr  = exp2f(m_run - m_new);
        float p0[8], p1[8], lsum = 0.0f;
        #pragma unroll
        for (int r = 0; r < 8; ++r) {
            p0[r] = exp2f(s0[r] - m_new);
            p1[r] = exp2f(s1[r] - m_new);
            lsum += p0[r] + p1[r];
        }
        l_run = l_run * corr + lsum;
        m_run = m_new;
        v8u pv;
        #pragma unroll
        for (int w = 0; w < 4; ++w) {
            pv[w]     = pk2bf(p0[2 * w], p0[2 * w + 1]);
            pv[4 + w] = pk2bf(p1[2 * w], p1[2 * w + 1]);
        }
        v16bf pF = __builtin_bit_cast(v16bf, pv);

        const int srcl = 8 * hi;
        #pragma unroll
        for (int r = 0; r < 8; ++r) {
            float cr = __shfl(corr, srcl + r, 32);
            o0[r] *= cr; o1[r] *= cr; o2[r] *= cr; o3[r] *= cr;
        }
        #pragma unroll
        for (int nb = 0; nb < 4; ++nb) {
            const float* vcol = vb + (size_t)(sb + 8 * hi) * ROW + nb * 16 + l15;
            v8u tw;
            #pragma unroll
            for (int w = 0; w < 8; ++w)
                tw[w] = pk2bf(vcol[(size_t)(2 * w + 8 * (w >> 2)) * ROW], vcol[(size_t)(2 * w + 8 * (w >> 2) + 1) * ROW]);
            v16bf vF = __builtin_bit_cast(v16bf, tw);
            v8f& oo = (nb == 0) ? o0 : (nb == 1) ? o1 : (nb == 2) ? o2 : o3;
            oo = __builtin_amdgcn_wmma_f32_16x16x32_bf16(
                     false, pF, false, vF, (short)0, oo, false, false);
        }
    }

    float lt = l_run + __shfl_xor(l_run, 16, 32);
    const int srcl = 8 * hi;
    #pragma unroll
    for (int r = 0; r < 8; ++r) {
        float inv = 1.0f / __shfl(lt, srcl + r, 32);
        float* orow = out + ((size_t)((n * L_ + qbase + r + 8 * hi) * H_) + h) * D_ + l15;
        orow[0]  = o0[r] * inv;
        orow[16] = o1[r] * inv;
        orow[32] = o2[r] * inv;
        orow[48] = o3[r] * inv;
    }
}

extern "C" void kernel_launch(void* const* d_in, const int* in_sizes, int n_in,
                              void* d_out, int out_size, void* d_ws, size_t ws_size,
                              hipStream_t stream) {
    (void)in_sizes; (void)n_in; (void)out_size;
    const float* qf = (const float*)d_in[0];
    const float* kf = (const float*)d_in[1];
    const float* vf = (const float*)d_in[2];
    const unsigned char* kvm = (const unsigned char*)d_in[4];
    float* out = (float*)d_out;

    if (ws_size >= WS_NEEDED && d_ws != nullptr) {
        uint32_t* kbf = (uint32_t*)d_ws;
        uint32_t* vtb = (uint32_t*)((char*)d_ws + KBF_BYTES);
        conv_k_bf16 <<<(N_ * S_ * H_ * D_ / 4) / 256, 256, 0, stream>>>(kf, kbf);
        conv_v_bf16T<<<(N_ * H_ * D_ * S_ / 2) / 256, 256, 0, stream>>>(vf, vtb);
        fattn_bf16  <<<(N_ * H_ * (L_ / 32)) / WAVES, WAVES * 32, 0, stream>>>(
            qf, (const short*)kbf, (const short*)vtb, kvm, out);
    } else {
        fattn_fused <<<(N_ * H_ * (L_ / 16)) / WAVES, WAVES * 32, 0, stream>>>(
            qf, kf, vf, kvm, out);
    }
}
